// CA_80461917323389
// MI455X (gfx1250) — matrix operand order repaired and checked by reading, not compiled, not run on hardware
//
#include <hip/hip_runtime.h>

#define B_   8
#define T_   8192
#define N_   256
#define D_   512
#define TXT_ 768
#define TE_  2048
#define H_   8
#define DH_  64

typedef __attribute__((ext_vector_type(16))) __bf16 v16bf;
typedef __attribute__((ext_vector_type(8)))  float  v8f;

struct U16x16 { uint4 lo, hi; };
union BfPack16 { __bf16 h[16]; uint4 q[2]; };

__device__ __forceinline__ __bf16 f2bf(float f) {
  unsigned u = __builtin_bit_cast(unsigned, f);
  u += 0x7FFFu + ((u >> 16) & 1u);                 // round-to-nearest-even
  unsigned short h = (unsigned short)(u >> 16);
  return __builtin_bit_cast(__bf16, h);
}
__device__ __forceinline__ float bf2f(__bf16 b) {
  unsigned short s = __builtin_bit_cast(unsigned short, b);
  unsigned u = ((unsigned)s) << 16;
  return __builtin_bit_cast(float, u);
}

// A-fragment: 16x32 bf16, source laid out [m][k] with unit-stride k.
__device__ __forceinline__ v16bf load_a(const __bf16* base, int stride, int m0, int k0) {
  const int lane = threadIdx.x & 31;
  const __bf16* p = base + (size_t)(m0 + (lane & 15)) * stride + k0 + ((lane >> 4) << 3);
  U16x16 u;
  u.lo = *reinterpret_cast<const uint4*>(p);
  u.hi = *reinterpret_cast<const uint4*>(p + 16);
  return __builtin_bit_cast(v16bf, u);
}
// B-fragment: 32x16 bf16, source laid out TRANSPOSED [n][k] with unit-stride k.
__device__ __forceinline__ v16bf load_b(const __bf16* base, int stride, int n0, int k0) {
  const int lane = threadIdx.x & 31;
  const __bf16* p = base + (size_t)(n0 + (lane & 15)) * stride + k0 + ((lane >> 4) << 3);
  U16x16 u;
  u.lo = *reinterpret_cast<const uint4*>(p);
  u.hi = *reinterpret_cast<const uint4*>(p + 16);
  return __builtin_bit_cast(v16bf, u);
}

__device__ __forceinline__ v8f wmma_bf16(v16bf a, v16bf b, v8f c) {
  return __builtin_amdgcn_wmma_f32_16x16x32_bf16(false, a, false, b, (short)0, c, false, false);
}

__device__ __forceinline__ float wave_sum(float v) {
  #pragma unroll
  for (int m = 16; m >= 1; m >>= 1) v += __shfl_xor(v, m, 32);
  return v;
}
__device__ __forceinline__ float half_sum(float v) {   // within 16-lane group
  #pragma unroll
  for (int m = 8; m >= 1; m >>= 1) v += __shfl_xor(v, m, 32);
  return v;
}
__device__ __forceinline__ float half_max(float v) {
  #pragma unroll
  for (int m = 8; m >= 1; m >>= 1) v = fmaxf(v, __shfl_xor(v, m, 32));
  return v;
}

// ---- CDNA5 async memory->LDS tile copy (GLOBAL_LOAD_ASYNC_TO_LDS_B128, ASYNCcnt) ----
// GV mode: per enabled lane, copies 16 bytes from global vaddr to LDS address vDst. Complete with s_wait_asynccnt.
__device__ __forceinline__ void async_tile_load_256t(void* lds_dst, const void* gsrc, int bytes) {
  const unsigned lds_base = (unsigned)(unsigned long long)lds_dst;  // low 32 bits = LDS offset
  const char* g = (const char*)gsrc;
  for (int off = (int)threadIdx.x * 16; off < bytes; off += 256 * 16) {
    const unsigned ldst = lds_base + (unsigned)off;
    const char* gp = g + off;
    asm volatile("global_load_async_to_lds_b128 %0, %1, off"
                 :: "v"(ldst), "v"(gp) : "memory");
  }
  asm volatile("s_wait_asynccnt 0x0" ::: "memory");
}

// ---------------- weight convert + transpose: fp32 [K][N] -> bf16 [N][K] ----------------
__global__ void k_convT(const float* __restrict__ src, __bf16* __restrict__ dst, int K, int N) {
  int i = blockIdx.x * blockDim.x + threadIdx.x;
  if (i >= K * N) return;
  int k = i / N, n = i - k * N;
  dst[(size_t)n * K + k] = f2bf(src[i]);
}

// ---------------- LN(xf) -> bf16 XT [B*N][TXT]; one wave/row, 24 contiguous ch/lane ----------------
__global__ void k_ln_xf(const float* __restrict__ xf, const float* __restrict__ g,
                        const float* __restrict__ bta, __bf16* __restrict__ XT) {
  const int wave = threadIdx.x >> 5, lane = threadIdx.x & 31;
  const int row = blockIdx.x * 8 + wave;                  // < B_*N_
  const int c0 = lane * 24;
  const float* src = xf + (size_t)row * TXT_ + c0;
  float xv[24]; float s = 0.f;
  #pragma unroll
  for (int i = 0; i < 24; ++i) { xv[i] = src[i]; s += xv[i]; }
  s = wave_sum(s);
  const float mean = s * (1.f / TXT_);
  float vs = 0.f;
  #pragma unroll
  for (int i = 0; i < 24; ++i) { float d = xv[i] - mean; vs += d * d; }
  vs = wave_sum(vs);
  const float inv = rsqrtf(vs * (1.f / TXT_) + 1e-5f);
  union { __bf16 h[24]; uint4 q[3]; } u;
  #pragma unroll
  for (int i = 0; i < 24; ++i) u.h[i] = f2bf((xv[i] - mean) * inv * g[c0 + i] + bta[c0 + i]);
  uint4* dst = reinterpret_cast<uint4*>(XT + (size_t)row * TXT_ + c0);   // 48B/lane, 16B aligned
  dst[0] = u.q[0]; dst[1] = u.q[1]; dst[2] = u.q[2];
}

// ---------------- scale/shift = silu(emb) @ emb_W + emb_b : [B][2D] ----------------
__global__ void k_emb(const float* __restrict__ emb, const float* __restrict__ W,
                      const float* __restrict__ bias, float* __restrict__ SS) {
  const int b = blockIdx.y;
  const int col = blockIdx.x * 256 + threadIdx.x;          // < 2*D_
  float acc = bias[col];
  const float* e = emb + (size_t)b * TE_;
  for (int k = 0; k < TE_; ++k) {
    const float xk = e[k];
    acc += (xk / (1.f + __expf(-xk))) * W[(size_t)k * (2 * D_) + col];
  }
  SS[(size_t)b * (2 * D_) + col] = acc;
}

// ---------------- k = softmax(xt@Wk+bk), v = xt@Wv+bv ; stored TRANSPOSED [B][D][N] ----------------
__global__ __launch_bounds__(256) void k_kvproj(
    const __bf16* __restrict__ XT, const __bf16* __restrict__ WkT, const __bf16* __restrict__ WvT,
    const float* __restrict__ bk, const float* __restrict__ bv,
    __bf16* __restrict__ KT, __bf16* __restrict__ VT) {
  __shared__ __bf16 s_t[64 * TXT_];                        // 96 KB
  const int z = blockIdx.z;                                // 0 = K (softmax), 1 = V
  const __bf16* WT  = z ? WvT : WkT;
  const float*  bias = z ? bv  : bk;
  __bf16*       OT  = z ? VT  : KT;
  const int b = blockIdx.y;
  const int m0 = blockIdx.x * 64;
  async_tile_load_256t(s_t, XT + ((size_t)b * N_ + m0) * TXT_, 64 * TXT_ * 2);
  __syncthreads();
  const int wave = threadIdx.x >> 5, lane = threadIdx.x & 31;
  const int laneL = lane & 15, ro = (lane >> 4) * 8;
  const int nstrip = wave * 64;                            // wave == head
  v8f acc[4][4] = {};
  for (int kt = 0; kt < TXT_ / 32; ++kt) {
    v16bf bf[4];
    #pragma unroll
    for (int ni = 0; ni < 4; ++ni) bf[ni] = load_b(WT, TXT_, nstrip + 16 * ni, 32 * kt);
    #pragma unroll
    for (int mi = 0; mi < 4; ++mi) {
      const v16bf af = load_a(s_t, TXT_, 16 * mi, 32 * kt);
      #pragma unroll
      for (int ni = 0; ni < 4; ++ni) acc[mi][ni] = wmma_bf16(af, bf[ni], acc[mi][ni]);
    }
  }
  #pragma unroll
  for (int mi = 0; mi < 4; ++mi)
    #pragma unroll
    for (int ni = 0; ni < 4; ++ni) {
      const float bb = bias[nstrip + 16 * ni + laneL];
      #pragma unroll
      for (int r = 0; r < 8; ++r) acc[mi][ni][r] += bb;
    }
  if (z == 0) {                                            // per-head softmax over DH=64
    #pragma unroll
    for (int mi = 0; mi < 4; ++mi)
      #pragma unroll
      for (int r = 0; r < 8; ++r) {
        float mx = -3.0e38f;
        #pragma unroll
        for (int ni = 0; ni < 4; ++ni) mx = fmaxf(mx, acc[mi][ni][r]);
        mx = half_max(mx);
        float ev[4], s = 0.f;
        #pragma unroll
        for (int ni = 0; ni < 4; ++ni) { ev[ni] = __expf(acc[mi][ni][r] - mx); s += ev[ni]; }
        s = half_sum(s);
        const float invs = 1.f / s;
        #pragma unroll
        for (int ni = 0; ni < 4; ++ni) acc[mi][ni][r] = ev[ni] * invs;
      }
  }
  #pragma unroll
  for (int mi = 0; mi < 4; ++mi)
    #pragma unroll
    for (int ni = 0; ni < 4; ++ni) {
      const int col = nstrip + 16 * ni + laneL;
      #pragma unroll
      for (int r = 0; r < 8; ++r) {
        const int n = m0 + 16 * mi + ro + r;
        OT[((size_t)b * D_ + col) * N_ + n] = f2bf(acc[mi][ni][r]);
      }
    }
}

// ---------------- attn[b][h] = k^T v : 64x64x256 GEMM; stored TRANSPOSED [l][d] ----------------
__global__ __launch_bounds__(128) void k_attn(const __bf16* __restrict__ KT,
                                              const __bf16* __restrict__ VT,
                                              __bf16* __restrict__ ATT) {
  const int h = blockIdx.x, b = blockIdx.y;
  const int wave = threadIdx.x >> 5, lane = threadIdx.x & 31;
  const int laneL = lane & 15, ro = (lane >> 4) * 8;
  const __bf16* kbase = KT + ((size_t)b * D_ + h * 64) * N_;   // [d][n]
  const __bf16* vbase = VT + ((size_t)b * D_ + h * 64) * N_;   // [l][n]
  v8f acc[4] = {};
  for (int kt = 0; kt < N_ / 32; ++kt) {
    const v16bf af = load_a(kbase, N_, 16 * wave, 32 * kt);    // A[d][n]
    #pragma unroll
    for (int ni = 0; ni < 4; ++ni) {
      const v16bf bf = load_b(vbase, N_, 16 * ni, 32 * kt);    // B[n][l] via [l][n]
      acc[ni] = wmma_bf16(af, bf, acc[ni]);
    }
  }
  __bf16* at = ATT + (size_t)(b * H_ + h) * 64 * 64;           // attnT [l][d]
  #pragma unroll
  for (int ni = 0; ni < 4; ++ni)
    #pragma unroll
    for (int r = 0; r < 8; ++r) {
      const int l = 16 * ni + laneL;
      const int d = 16 * wave + ro + r;
      at[l * 64 + d] = f2bf(acc[ni][r]);
    }
}

// ---------------- fused: LN(x) -> q-proj -> softmax -> q@attn -> stylize/silu -> HB ----------------
__global__ __launch_bounds__(256) void k_main(
    const float* __restrict__ x, const float* __restrict__ g_lnx, const float* __restrict__ b_lnx,
    const __bf16* __restrict__ WqT, const float* __restrict__ bq,
    const __bf16* __restrict__ ATT, const float* __restrict__ SS,
    const float* __restrict__ g_lno, const float* __restrict__ b_lno,
    __bf16* __restrict__ HB) {
  __shared__ __bf16 s_a[64 * D_];                          // LN(x) tile, later reused for 'out'
  __shared__ __bf16 s_q[64 * D_];                          // softmaxed q
  const int b = blockIdx.y;
  const int m0 = blockIdx.x * 64;
  const int wave = threadIdx.x >> 5, lane = threadIdx.x & 31;
  const int laneL = lane & 15, ro = (lane >> 4) * 8;
  const int nstrip = wave * 64;                            // wave == head
  const int c0 = lane * 16;                                // 16 contiguous channels per lane

  // stage 1: LayerNorm(x) -> s_a (bf16), fully vectorized b128 path
  for (int rr = 0; rr < 8; ++rr) {
    const int row = wave * 8 + rr;
    const float* src = x + ((size_t)b * T_ + m0 + row) * D_ + c0;
    float xv[16]; float s = 0.f;
    #pragma unroll
    for (int i = 0; i < 16; ++i) { xv[i] = src[i]; s += xv[i]; }
    s = wave_sum(s);
    const float mean = s * (1.f / D_);
    float vs = 0.f;
    #pragma unroll
    for (int i = 0; i < 16; ++i) { float d = xv[i] - mean; vs += d * d; }
    vs = wave_sum(vs);
    const float inv = rsqrtf(vs * (1.f / D_) + 1e-5f);
    BfPack16 u;
    #pragma unroll
    for (int i = 0; i < 16; ++i)
      u.h[i] = f2bf((xv[i] - mean) * inv * g_lnx[c0 + i] + b_lnx[c0 + i]);
    uint4* dst = reinterpret_cast<uint4*>(s_a + row * D_ + c0);
    dst[0] = u.q[0]; dst[1] = u.q[1];
  }
  __syncthreads();

  // stage 2: qraw strip = s_a @ Wq (K=512)
  v8f acc[4][4] = {};
  for (int kt = 0; kt < D_ / 32; ++kt) {
    v16bf bf[4];
    #pragma unroll
    for (int ni = 0; ni < 4; ++ni) bf[ni] = load_b(WqT, D_, nstrip + 16 * ni, 32 * kt);
    #pragma unroll
    for (int mi = 0; mi < 4; ++mi) {
      const v16bf af = load_a(s_a, D_, 16 * mi, 32 * kt);
      #pragma unroll
      for (int ni = 0; ni < 4; ++ni) acc[mi][ni] = wmma_bf16(af, bf[ni], acc[mi][ni]);
    }
  }
  __syncthreads();                                         // everyone done reading s_a

  // stage 3: +bq, per-head softmax, q -> s_q
  #pragma unroll
  for (int mi = 0; mi < 4; ++mi)
    #pragma unroll
    for (int ni = 0; ni < 4; ++ni) {
      const float bb = bq[nstrip + 16 * ni + laneL];
      #pragma unroll
      for (int r = 0; r < 8; ++r) acc[mi][ni][r] += bb;
    }
  #pragma unroll
  for (int mi = 0; mi < 4; ++mi)
    #pragma unroll
    for (int r = 0; r < 8; ++r) {
      float mx = -3.0e38f;
      #pragma unroll
      for (int ni = 0; ni < 4; ++ni) mx = fmaxf(mx, acc[mi][ni][r]);
      mx = half_max(mx);
      float ev[4], s = 0.f;
      #pragma unroll
      for (int ni = 0; ni < 4; ++ni) { ev[ni] = __expf(acc[mi][ni][r] - mx); s += ev[ni]; }
      s = half_sum(s);
      const float invs = 1.f / s;
      const int row = 16 * mi + ro + r;
      #pragma unroll
      for (int ni = 0; ni < 4; ++ni)
        s_q[row * D_ + nstrip + 16 * ni + laneL] = f2bf(ev[ni] * invs);
    }

  // stage 4: out_head = q_head @ attn[b][head] (64x64x64), out -> s_a (bf16)
  #pragma unroll
  for (int mi = 0; mi < 4; ++mi)
    #pragma unroll
    for (int ni = 0; ni < 4; ++ni)
      #pragma unroll
      for (int r = 0; r < 8; ++r) acc[mi][ni][r] = 0.f;
  const __bf16* at = ATT + (size_t)(b * H_ + wave) * 64 * 64;  // [l][d]
  #pragma unroll
  for (int kt = 0; kt < 2; ++kt) {
    v16bf bf[4];
    #pragma unroll
    for (int ni = 0; ni < 4; ++ni) bf[ni] = load_b(at, 64, 16 * ni, 32 * kt);
    #pragma unroll
    for (int mi = 0; mi < 4; ++mi) {
      const v16bf af = load_a(s_q + nstrip, D_, 16 * mi, 32 * kt);
      #pragma unroll
      for (int ni = 0; ni < 4; ++ni) acc[mi][ni] = wmma_bf16(af, bf[ni], acc[mi][ni]);
    }
  }
  #pragma unroll
  for (int mi = 0; mi < 4; ++mi)
    #pragma unroll
    for (int ni = 0; ni < 4; ++ni)
      #pragma unroll
      for (int r = 0; r < 8; ++r)
        s_a[(16 * mi + ro + r) * D_ + nstrip + 16 * ni + laneL] = f2bf(acc[mi][ni][r]);
  __syncthreads();

  // stage 5: h = silu(LN(out)*(1+scale)+shift) -> HB (bf16), b128 in/out
  const float* scale = SS + (size_t)b * (2 * D_);
  const float* shift = scale + D_;
  for (int rr = 0; rr < 8; ++rr) {
    const int row = wave * 8 + rr;
    BfPack16 uin;
    const uint4* sp = reinterpret_cast<const uint4*>(s_a + row * D_ + c0);
    uin.q[0] = sp[0]; uin.q[1] = sp[1];
    float xv[16]; float s = 0.f;
    #pragma unroll
    for (int i = 0; i < 16; ++i) { xv[i] = bf2f(uin.h[i]); s += xv[i]; }
    s = wave_sum(s);
    const float mean = s * (1.f / D_);
    float vs = 0.f;
    #pragma unroll
    for (int i = 0; i < 16; ++i) { float d = xv[i] - mean; vs += d * d; }
    vs = wave_sum(vs);
    const float inv = rsqrtf(vs * (1.f / D_) + 1e-5f);
    BfPack16 uo;
    #pragma unroll
    for (int i = 0; i < 16; ++i) {
      const int c = c0 + i;
      float o = (xv[i] - mean) * inv * g_lno[c] + b_lno[c];
      o = o * (1.f + scale[c]) + shift[c];
      o = o / (1.f + __expf(-o));                          // silu
      uo.h[i] = f2bf(o);
    }
    uint4* dst = reinterpret_cast<uint4*>(HB + ((size_t)b * T_ + m0 + row) * D_ + c0);
    dst[0] = uo.q[0]; dst[1] = uo.q[1];
  }
}

// ---------------- y = x + h @ out_W + out_b ----------------
__global__ __launch_bounds__(256) void k_final(
    const __bf16* __restrict__ HB, const __bf16* __restrict__ WoT,
    const float* __restrict__ ob, const float* __restrict__ x, float* __restrict__ y) {
  __shared__ __bf16 s_h[64 * D_];
  const int b = blockIdx.y;
  const int m0 = blockIdx.x * 64;
  const int wave = threadIdx.x >> 5, lane = threadIdx.x & 31;
  const int laneL = lane & 15, ro = (lane >> 4) * 8;
  const int nstrip = wave * 64;
  async_tile_load_256t(s_h, HB + ((size_t)b * T_ + m0) * D_, 64 * D_ * 2);
  __syncthreads();
  v8f acc[4][4] = {};
  for (int kt = 0; kt < D_ / 32; ++kt) {
    v16bf bf[4];
    #pragma unroll
    for (int ni = 0; ni < 4; ++ni) bf[ni] = load_b(WoT, D_, nstrip + 16 * ni, 32 * kt);
    #pragma unroll
    for (int mi = 0; mi < 4; ++mi) {
      const v16bf af = load_a(s_h, D_, 16 * mi, 32 * kt);
      #pragma unroll
      for (int ni = 0; ni < 4; ++ni) acc[mi][ni] = wmma_bf16(af, bf[ni], acc[mi][ni]);
    }
  }
  #pragma unroll
  for (int mi = 0; mi < 4; ++mi)
    #pragma unroll
    for (int ni = 0; ni < 4; ++ni) {
      const int col = nstrip + 16 * ni + laneL;
      const float bb = ob[col];
      #pragma unroll
      for (int r = 0; r < 8; ++r) {
        const size_t idx = ((size_t)b * T_ + m0 + 16 * mi + ro + r) * D_ + col;
        y[idx] = acc[mi][ni][r] + bb + x[idx];
      }
    }
}

extern "C" void kernel_launch(void* const* d_in, const int* in_sizes, int n_in,
                              void* d_out, int out_size, void* d_ws, size_t ws_size,
                              hipStream_t stream) {
  const float* x     = (const float*)d_in[0];
  const float* xf    = (const float*)d_in[1];
  const float* emb   = (const float*)d_in[2];
  const float* lnx_g = (const float*)d_in[3];
  const float* lnx_b = (const float*)d_in[4];
  const float* lnt_g = (const float*)d_in[5];
  const float* lnt_b = (const float*)d_in[6];
  const float* Wq    = (const float*)d_in[7];
  const float* bq    = (const float*)d_in[8];
  const float* Wk    = (const float*)d_in[9];
  const float* bk    = (const float*)d_in[10];
  const float* Wv    = (const float*)d_in[11];
  const float* bv    = (const float*)d_in[12];
  const float* embW  = (const float*)d_in[13];
  const float* embB  = (const float*)d_in[14];
  const float* lno_g = (const float*)d_in[15];
  const float* lno_b = (const float*)d_in[16];
  const float* Wo    = (const float*)d_in[17];
  const float* ob    = (const float*)d_in[18];

  // workspace layout (bf16 region then fp32 scale/shift); total ~78 MB
  __bf16* p = (__bf16*)d_ws;
  __bf16* WqT = p; p += (size_t)D_ * D_;
  __bf16* WkT = p; p += (size_t)TXT_ * D_;
  __bf16* WvT = p; p += (size_t)TXT_ * D_;
  __bf16* WoT = p; p += (size_t)D_ * D_;
  __bf16* XT  = p; p += (size_t)B_ * N_ * TXT_;
  __bf16* KT  = p; p += (size_t)B_ * D_ * N_;
  __bf16* VT  = p; p += (size_t)B_ * D_ * N_;
  __bf16* ATT = p; p += (size_t)B_ * H_ * DH_ * DH_;
  __bf16* HB  = p; p += (size_t)B_ * T_ * D_;
  float*  SS  = (float*)p;

  k_convT<<<(D_ * D_ + 255) / 256, 256, 0, stream>>>(Wq, WqT, D_, D_);
  k_convT<<<(TXT_ * D_ + 255) / 256, 256, 0, stream>>>(Wk, WkT, TXT_, D_);
  k_convT<<<(TXT_ * D_ + 255) / 256, 256, 0, stream>>>(Wv, WvT, TXT_, D_);
  k_convT<<<(D_ * D_ + 255) / 256, 256, 0, stream>>>(Wo, WoT, D_, D_);
  k_ln_xf<<<B_ * N_ / 8, 256, 0, stream>>>(xf, lnt_g, lnt_b, XT);
  k_emb<<<dim3((2 * D_) / 256, B_), 256, 0, stream>>>(emb, embW, embB, SS);
  k_kvproj<<<dim3(N_ / 64, B_, 2), 256, 0, stream>>>(XT, WkT, WvT, bk, bv, KT, VT);
  k_attn<<<dim3(H_, B_), 128, 0, stream>>>(KT, VT, ATT);
  k_main<<<dim3(T_ / 64, B_), 256, 0, stream>>>(x, lnx_g, lnx_b, WqT, bq, ATT, SS,
                                                lno_g, lno_b, HB);
  k_final<<<dim3(T_ / 64, B_), 256, 0, stream>>>(HB, WoT, ob, x, (float*)d_out);

  (void)in_sizes; (void)n_in; (void)out_size; (void)ws_size;
}
